// SceneGraphTransformer_60138132078621
// MI455X (gfx1250) — compile-verified, matrix operand order checked by reading, not run on hardware
//
#include <hip/hip_runtime.h>

// ---------------------------------------------------------------------------
// SceneGraphTransformer forward for MI455X (gfx1250, wave32, WMMA bf16)
// B=8 N=128 D=256 H=8 Hd=32 L=4 FF=1024 NOBJ=150 NREL=51
// ---------------------------------------------------------------------------

#define Bc 8
#define Nc 128
#define Dc 256
#define Hc 8
#define Hd 32
#define Lc 4
#define FFc 1024
#define NOBJ 150
#define NRELm1 50

typedef __attribute__((ext_vector_type(16))) __bf16 v16bf;
typedef __attribute__((ext_vector_type(8)))  float  v8f;

union Frag {
    v16bf v;
    unsigned int u[8];
};

__device__ __forceinline__ unsigned short f2bf(float f) {
    unsigned int u = __float_as_uint(f);
    unsigned int r = u + 0x7FFFu + ((u >> 16) & 1u);
    return (unsigned short)(r >> 16);
}

__device__ __forceinline__ void load_frag_lds(Frag& a, const unsigned short* ap) {
    // ap must be 16-byte aligned; two ds_load_b128
    uint4 t0 = *(const uint4*)ap;
    uint4 t1 = *(const uint4*)(ap + 16);
    a.u[0] = t0.x; a.u[1] = t0.y; a.u[2] = t0.z; a.u[3] = t0.w;
    a.u[4] = t1.x; a.u[5] = t1.y; a.u[6] = t1.z; a.u[7] = t1.w;
}

__device__ __forceinline__ void load_frag_glb(Frag& a, const unsigned int* gp) {
    // gp is 32-byte aligned; two global_load_b128
    uint4 t0 = *(const uint4*)gp;
    uint4 t1 = *(const uint4*)(gp + 4);
    a.u[0] = t0.x; a.u[1] = t0.y; a.u[2] = t0.z; a.u[3] = t0.w;
    a.u[4] = t1.x; a.u[5] = t1.y; a.u[6] = t1.z; a.u[7] = t1.w;
}

__device__ __forceinline__ float gelu_exact(float x) {
    return 0.5f * x * (1.0f + erff(x * 0.70710678118654752f));
}

// ---------------------------------------------------------------------------
// Weight prep: f32 [K, NOUT] row-major -> bf16 WMMA B-fragments.
// Fragment f = kb*ntiles + nt occupies 256 dwords; dword index within frag
// = lane*8 + v. lane: n = nt*16 + (lane&15), half = lane>>4.
// v: k = kb*32 + (v>>2)*16 + half*8 + (v&3)*2  (pair k, k+1 packed lo/hi).
// Columns >= NOUT are zero-padded.
// ---------------------------------------------------------------------------
__global__ void prep_frag_kernel(const float* __restrict__ w,
                                 unsigned int* __restrict__ f,
                                 int K, int NOUT, int ntiles) {
    int tid = blockIdx.x * 256 + threadIdx.x;
    int total = (K >> 5) * ntiles * 256;
    if (tid >= total) return;
    int fr = tid >> 8;
    int r  = tid & 255;
    int l  = r >> 3;
    int v  = r & 7;
    int kb = fr / ntiles;
    int nt = fr % ntiles;
    int n  = (nt << 4) + (l & 15);
    int hl = l >> 4;
    int k  = (kb << 5) + ((v >> 2) << 4) + (hl << 3) + ((v & 3) << 1);
    unsigned short lo = 0, hi = 0;
    if (n < NOUT) {
        lo = f2bf(w[(size_t)k * NOUT + n]);
        hi = f2bf(w[(size_t)(k + 1) * NOUT + n]);
    }
    f[tid] = (unsigned int)lo | ((unsigned int)hi << 16);
}

// ---------------------------------------------------------------------------
// Generic GEMM: out[M,NOUT] = act(A[M,K] @ W + bias) (+resid). bf16 WMMA.
// block = 128 threads (4 waves). grid = (M/16, ceil(ntiles/4)).
// A is f32 row-major, staged to LDS as bf16. K in {256,1024} (pass lgK).
// Inner loop is software-pipelined one k-step deep so each WMMA overlaps the
// next fragment fetch (partial s_wait instead of full drain).
// ---------------------------------------------------------------------------
__global__ void gemm_bf16_kernel(const float* __restrict__ A,
                                 const unsigned int* __restrict__ Bf,
                                 const float* __restrict__ bias,
                                 float* __restrict__ out,
                                 const float* __restrict__ resid,
                                 int K, int lgK, int NOUT, int ntiles, int act) {
    __shared__ unsigned short As[16 * 1024];
    unsigned int* Asu = (unsigned int*)As;
    const int tid = threadIdx.x;
    const int m0  = blockIdx.x << 4;
    const int lgKd = lgK - 1;                 // log2(dwords per row)
    const int Kdm1 = (1 << lgKd) - 1;
    // stage A tile as bf16 pairs (float2 global loads, dword LDS stores)
    for (int idx = tid; idx < (16 << lgKd); idx += 128) {
        int m  = idx >> lgKd;
        int kd = idx & Kdm1;
        float2 xv = *(const float2*)(A + ((size_t)(m0 + m) << lgK) + (kd << 1));
        Asu[(m << lgKd) + kd] = (unsigned int)f2bf(xv.x) | ((unsigned int)f2bf(xv.y) << 16);
    }
    __syncthreads();
    const int wid  = tid >> 5;
    const int lane = tid & 31;
    const int nt   = blockIdx.y * 4 + wid;
    if (nt >= ntiles) return;
    const int mrow = lane & 15;
    const int hl   = lane >> 4;
    v8f acc = {0.f, 0.f, 0.f, 0.f, 0.f, 0.f, 0.f, 0.f};
    const int ksteps = K >> 5;
    const unsigned short* arow = As + (mrow << lgK) + (hl << 3);
    const unsigned int*   brow = Bf + ((size_t)nt << 8) + (lane << 3);
    const size_t bstride = (size_t)ntiles << 8;
    Frag a_cur, b_cur;
    load_frag_lds(a_cur, arow);
    load_frag_glb(b_cur, brow);
    for (int kb = 0; kb < ksteps - 1; ++kb) {
        Frag a_nxt, b_nxt;
        load_frag_lds(a_nxt, arow + ((kb + 1) << 5));
        load_frag_glb(b_nxt, brow + (size_t)(kb + 1) * bstride);
        acc = __builtin_amdgcn_wmma_f32_16x16x32_bf16(false, a_cur.v, false, b_cur.v,
                                                      (short)0, acc, false, false);
        a_cur = a_nxt;
        b_cur = b_nxt;
    }
    acc = __builtin_amdgcn_wmma_f32_16x16x32_bf16(false, a_cur.v, false, b_cur.v,
                                                  (short)0, acc, false, false);
    const int n = (nt << 4) + mrow;
    if (n < NOUT) {
        const float bv = bias[n];
#pragma unroll
        for (int r = 0; r < 8; ++r) {
            int m = m0 + (hl << 3) + r;
            float val = acc[r] + bv;
            if (act == 1) val = gelu_exact(val);
            if (resid) val += resid[(size_t)m * NOUT + n];
            out[(size_t)m * NOUT + n] = val;
        }
    }
}

// ---------------------------------------------------------------------------
// Time-embedding MLP: one block per batch element.
// ---------------------------------------------------------------------------
__global__ void time_mlp_kernel(const int* __restrict__ t,
                                const float* __restrict__ w1, const float* __restrict__ b1,
                                const float* __restrict__ w2, const float* __restrict__ b2,
                                float* __restrict__ temb) {
    __shared__ float te0[256];
    __shared__ float te1[256];
    const int b = blockIdx.x, c = threadIdx.x;
    const float tv = (float)t[b];
    const float lg = logf(10000.0f);
    if (c < 128) te0[c] = cosf(tv * __expf(-lg * (float)c / 128.0f));
    else         te0[c] = sinf(tv * __expf(-lg * (float)(c - 128) / 128.0f));
    __syncthreads();
    float s = b1[c];
    for (int k = 0; k < 256; ++k) s += te0[k] * w1[k * 256 + c];
    te1[c] = s / (1.0f + __expf(-s));
    __syncthreads();
    float s2 = b2[c];
    for (int k = 0; k < 256; ++k) s2 += te1[k] * w2[k * 256 + c];
    temb[b * 256 + c] = s2;
}

// ---------------------------------------------------------------------------
// h = (obj_emb[obj_t] + temb) * node_mask
// ---------------------------------------------------------------------------
__global__ void embed_kernel(const int* __restrict__ obj_t,
                             const float* __restrict__ obj_emb,
                             const float* __restrict__ temb,
                             const int* __restrict__ node_mask,
                             float* __restrict__ h) {
    int tid = blockIdx.x * 256 + threadIdx.x;
    int bn = tid >> 8;
    int dd = tid & 255;
    int b = bn >> 7;
    int o = obj_t[bn];
    float v = obj_emb[(size_t)o * 256 + dd] + temb[b * 256 + dd];
    h[tid] = node_mask[bn] ? v : 0.0f;
}

__global__ void maskmul_kernel(float* __restrict__ h, const int* __restrict__ node_mask) {
    int tid = blockIdx.x * 256 + threadIdx.x;
    if (!node_mask[tid >> 8]) h[tid] = 0.0f;
}

// ---------------------------------------------------------------------------
// LayerNorm: one block (256 threads) per row of 256.
// ---------------------------------------------------------------------------
__global__ void ln_kernel(const float* __restrict__ in, float* __restrict__ out,
                          const float* __restrict__ g, const float* __restrict__ be) {
    __shared__ float red[256];
    const int row = blockIdx.x, tid = threadIdx.x;
    const float x = in[(size_t)row * 256 + tid];
    red[tid] = x;
    __syncthreads();
    for (int o = 128; o > 0; o >>= 1) { if (tid < o) red[tid] += red[tid + o]; __syncthreads(); }
    const float mean = red[0] * (1.0f / 256.0f);
    __syncthreads();
    const float d = x - mean;
    red[tid] = d * d;
    __syncthreads();
    for (int o = 128; o > 0; o >>= 1) { if (tid < o) red[tid] += red[tid + o]; __syncthreads(); }
    const float var = red[0] * (1.0f / 256.0f);
    out[(size_t)row * 256 + tid] = d * rsqrtf(var + 1e-5f) * g[tid] + be[tid];
}

// ---------------------------------------------------------------------------
// Attention bias: bias[b,h,i,j] = rel_emb[rel_t[b,i,j]] . eb_w[:,h] + eb_b[h]
// grid = B*N (one block per (b,i)), 128 threads (thread = j). Table is
// L2-resident (51 KB) so the "gather" never hits HBM.
// ---------------------------------------------------------------------------
__global__ void bias_kernel(const int* __restrict__ rel_t,
                            const float* __restrict__ rel_emb,
                            const float* __restrict__ ebw,
                            const float* __restrict__ ebb,
                            float* __restrict__ biasbuf) {
    const int bi = blockIdx.x;
    const int b = bi >> 7, i = bi & 127;
    const int j = threadIdx.x;
    const int r = rel_t[(size_t)bi * 128 + j];
    const float* e = rel_emb + (size_t)r * 256;
    float acc[8] = {0, 0, 0, 0, 0, 0, 0, 0};
    for (int dd = 0; dd < 256; ++dd) {
        float ev = e[dd];
        const float* w = ebw + dd * 8;
#pragma unroll
        for (int h = 0; h < 8; ++h) acc[h] += ev * w[h];
    }
#pragma unroll
    for (int h = 0; h < 8; ++h)
        biasbuf[((((size_t)b * 8 + h) * 128) + i) * 128 + j] = acc[h] + ebb[h];
}

// ---------------------------------------------------------------------------
// Attention core: grid = B*H*N (block per (b,h,i)), 128 threads (thread = j).
// ---------------------------------------------------------------------------
__global__ void attn_kernel(const float* __restrict__ q, const float* __restrict__ k,
                            const float* __restrict__ v, const float* __restrict__ biasbuf,
                            const int* __restrict__ node_mask, const int* __restrict__ edge_mask,
                            float* __restrict__ outp) {
    __shared__ float qs[32];
    __shared__ float lg[128];
    __shared__ float sred[2];
    const int blk = blockIdx.x;
    const int b = blk >> 10;
    const int h = (blk >> 7) & 7;
    const int i = blk & 127;
    const int tid = threadIdx.x;
    const size_t qoff = ((size_t)(b * 128 + i)) * 256 + h * 32;
    if (tid < 32) qs[tid] = q[qoff + tid];
    __syncthreads();
    const int j = tid;
    const float* kr = k + ((size_t)(b * 128 + j)) * 256 + h * 32;
    float dot = 0.0f;
#pragma unroll
    for (int d = 0; d < 32; ++d) dot += qs[d] * kr[d];
    float lo = dot * 0.17677669529663687f +
               biasbuf[((((size_t)b * 8 + h) * 128) + i) * 128 + j];
    const bool ok = (edge_mask[((size_t)(b * 128 + i)) * 128 + j] != 0) &&
                    (node_mask[b * 128 + j] != 0);
    lo = ok ? lo : -INFINITY;
    lg[j] = lo;
    __syncthreads();
    if (tid < 32) {
        float m = fmaxf(fmaxf(lg[tid], lg[tid + 32]), fmaxf(lg[tid + 64], lg[tid + 96]));
        for (int o = 16; o > 0; o >>= 1) m = fmaxf(m, __shfl_xor(m, o, 32));
        if (tid == 0) sred[0] = m;
    }
    __syncthreads();
    const float mx = sred[0];
    float e = (ok && mx > -INFINITY) ? __expf(lo - mx) : 0.0f;
    __syncthreads();
    lg[j] = e;
    __syncthreads();
    if (tid < 32) {
        float s = lg[tid] + lg[tid + 32] + lg[tid + 64] + lg[tid + 96];
        for (int o = 16; o > 0; o >>= 1) s += __shfl_xor(s, o, 32);
        if (tid == 0) sred[1] = s;
    }
    __syncthreads();
    if (tid < 32) {
        const float s = sred[1];
        const float inv = (s > 0.0f) ? 1.0f / s : 0.0f;
        const float* vb = v + ((size_t)b * 128) * 256 + h * 32 + tid;
        float a = 0.0f;
        for (int jj = 0; jj < 128; ++jj) a += lg[jj] * vb[(size_t)jj * 256];
        outp[qoff + tid] = a * inv;
    }
}

// ---------------------------------------------------------------------------
// Pair head: grid = B*N*(N/16) blocks, 32 threads (one wave). Each block:
// 16 pairs (b, i, j0..j0+15). The 16x1024 bf16 pair_feat tile
// [hi | hj | hi-hj | hi*hj] is built ONCE into LDS (featS, 32 KB), then the
// hot loop is 2x ds_load_b128 (A, double-buffered) + 4x global_load_b128 (B)
// + 2 WMMAs per k-step, dual-accumulating eh and rh. Second layers
// (256->1, 256->50) are scalar reductions from LDS.
// ---------------------------------------------------------------------------
__global__ void pair_head_kernel(const float* __restrict__ subjh,
                                 const float* __restrict__ objh,
                                 const unsigned int* __restrict__ ehf,
                                 const unsigned int* __restrict__ rhf,
                                 const float* __restrict__ ehb1,
                                 const float* __restrict__ rhb1,
                                 const float* __restrict__ ehw2,
                                 const float* __restrict__ ehb2,
                                 const float* __restrict__ rhw2,
                                 const float* __restrict__ rhb2,
                                 float* __restrict__ edge_out,
                                 float* __restrict__ rel_out) {
    __shared__ unsigned short featS[16 * 1024];   // 32 KB bf16 pair_feat tile
    __shared__ float actE[16][256];               // 16 KB
    __shared__ float actR[16][256];               // 16 KB
    unsigned int* featU = (unsigned int*)featS;
    const int blk = blockIdx.x;
    const int b  = blk >> 10;
    const int i  = (blk >> 3) & 127;
    const int jt = blk & 7;
    const int lane = threadIdx.x;
    const float* hrow = subjh + ((size_t)(b * 128 + i)) * 256;
    // ---- build feature tile: 16 rows x 512 dwords ----
    for (int idx = lane; idx < 16 * 512; idx += 32) {
        const int p  = idx >> 9;
        const int dw = idx & 511;
        const int k2 = dw << 1;
        const int seg = k2 >> 8;
        const int kk  = k2 & 255;
        const float* jrow = objh + ((size_t)(b * 128 + jt * 16 + p)) * 256;
        float2 hv = *(const float2*)(hrow + kk);
        float2 jv = *(const float2*)(jrow + kk);
        float a0 = (seg == 0) ? hv.x : (seg == 1) ? jv.x : (seg == 2) ? (hv.x - jv.x) : (hv.x * jv.x);
        float a1 = (seg == 0) ? hv.y : (seg == 1) ? jv.y : (seg == 2) ? (hv.y - jv.y) : (hv.y * jv.y);
        featU[idx] = (unsigned int)f2bf(a0) | ((unsigned int)f2bf(a1) << 16);
    }
    __syncthreads();
    const int mrow = lane & 15;
    const int hl   = lane >> 4;
    const unsigned short* arow = featS + (mrow << 10) + (hl << 3);
    for (int chunk = 0; chunk < 4; ++chunk) {
        v8f accE[4], accR[4];
#pragma unroll
        for (int c = 0; c < 4; ++c) {
            accE[c] = (v8f){0.f, 0.f, 0.f, 0.f, 0.f, 0.f, 0.f, 0.f};
            accR[c] = (v8f){0.f, 0.f, 0.f, 0.f, 0.f, 0.f, 0.f, 0.f};
        }
        Frag a_cur;
        load_frag_lds(a_cur, arow);
        for (int kb = 0; kb < 32; ++kb) {
            Frag a_nxt;
            if (kb < 31) load_frag_lds(a_nxt, arow + ((kb + 1) << 5));
#pragma unroll
            for (int c = 0; c < 4; ++c) {
                const int nt = chunk * 4 + c;
                Frag be, br;
                load_frag_glb(be, ehf + (((size_t)kb * 16 + nt) << 8) + (lane << 3));
                load_frag_glb(br, rhf + (((size_t)kb * 16 + nt) << 8) + (lane << 3));
                accE[c] = __builtin_amdgcn_wmma_f32_16x16x32_bf16(false, a_cur.v, false, be.v,
                                                                  (short)0, accE[c], false, false);
                accR[c] = __builtin_amdgcn_wmma_f32_16x16x32_bf16(false, a_cur.v, false, br.v,
                                                                  (short)0, accR[c], false, false);
            }
            a_cur = a_nxt;
        }
#pragma unroll
        for (int c = 0; c < 4; ++c) {
            const int nb = (chunk * 4 + c) * 16 + mrow;
            const float be1 = ehb1[nb], br1 = rhb1[nb];
#pragma unroll
            for (int r = 0; r < 8; ++r) {
                const int p = (hl << 3) + r;
                actE[p][nb] = gelu_exact(accE[c][r] + be1);
                actR[p][nb] = gelu_exact(accR[c][r] + br1);
            }
        }
    }
    __syncthreads();
    const size_t base = ((size_t)(b * 128 + i)) * 128 + jt * 16;
    for (int p = 0; p < 16; ++p) {
        float s = 0.0f;
        for (int c2 = lane; c2 < 256; c2 += 32) s += actE[p][c2] * ehw2[c2];
        for (int o = 16; o > 0; o >>= 1) s += __shfl_xor(s, o, 32);
        if (lane == 0) edge_out[base + p] = s + ehb2[0];
    }
    for (int idx = lane; idx < 16 * 50; idx += 32) {
        const int p = idx / 50, r = idx % 50;
        float s = rhb2[r];
        for (int c2 = 0; c2 < 256; ++c2) s += actR[p][c2] * rhw2[c2 * 50 + r];
        rel_out[(base + p) * 50 + r] = s;
    }
}

// ---------------------------------------------------------------------------
// Host launcher
// ---------------------------------------------------------------------------
extern "C" void kernel_launch(void* const* d_in, const int* in_sizes, int n_in,
                              void* d_out, int out_size, void* d_ws, size_t ws_size,
                              hipStream_t stream) {
    (void)in_sizes; (void)n_in; (void)out_size; (void)ws_size;
    const int*   obj_t      = (const int*)d_in[0];
    const int*   rel_t      = (const int*)d_in[1];
    const int*   t_in       = (const int*)d_in[2];
    const int*   node_mask  = (const int*)d_in[3];
    const int*   edge_mask  = (const int*)d_in[4];
    const float* obj_emb    = (const float*)d_in[5];
    const float* rel_emb    = (const float*)d_in[6];
    const float* time_w1    = (const float*)d_in[7];
    const float* time_b1    = (const float*)d_in[8];
    const float* time_w2    = (const float*)d_in[9];
    const float* time_b2    = (const float*)d_in[10];
    const float* ln1_g      = (const float*)d_in[11];
    const float* ln1_b      = (const float*)d_in[12];
    const float* q_w        = (const float*)d_in[13];
    const float* q_b        = (const float*)d_in[14];
    const float* k_w        = (const float*)d_in[15];
    const float* k_b        = (const float*)d_in[16];
    const float* v_w        = (const float*)d_in[17];
    const float* v_b        = (const float*)d_in[18];
    const float* o_w        = (const float*)d_in[19];
    const float* o_b        = (const float*)d_in[20];
    const float* eb_w       = (const float*)d_in[21];
    const float* eb_b       = (const float*)d_in[22];
    const float* ln2_g      = (const float*)d_in[23];
    const float* ln2_b      = (const float*)d_in[24];
    const float* ff_w1      = (const float*)d_in[25];
    const float* ff_b1      = (const float*)d_in[26];
    const float* ff_w2      = (const float*)d_in[27];
    const float* ff_b2      = (const float*)d_in[28];
    const float* obj_head_w = (const float*)d_in[29];
    const float* obj_head_b = (const float*)d_in[30];
    const float* subj_w     = (const float*)d_in[31];
    const float* subj_b     = (const float*)d_in[32];
    const float* objp_w     = (const float*)d_in[33];
    const float* objp_b     = (const float*)d_in[34];
    const float* eh_w1      = (const float*)d_in[35];
    const float* eh_b1      = (const float*)d_in[36];
    const float* eh_w2      = (const float*)d_in[37];
    const float* eh_b2      = (const float*)d_in[38];
    const float* rh_w1      = (const float*)d_in[39];
    const float* rh_b1      = (const float*)d_in[40];
    const float* rh_w2      = (const float*)d_in[41];
    const float* rh_b2      = (const float*)d_in[42];

    char* ws = (char*)d_ws;
    size_t cur = 0;
    auto alloc = [&](size_t bytes) -> void* {
        void* p = ws + cur;
        cur = (cur + bytes + 255) & ~(size_t)255;
        return p;
    };
    const size_t BN = (size_t)Bc * Nc;               // 1024
    float* temb    = (float*)alloc((size_t)Bc * Dc * 4);
    float* hbuf    = (float*)alloc(BN * Dc * 4);
    float* xbuf    = (float*)alloc(BN * Dc * 4);
    float* qbuf    = (float*)alloc(BN * Dc * 4);
    float* kbuf    = (float*)alloc(BN * Dc * 4);
    float* vbuf    = (float*)alloc(BN * Dc * 4);
    float* attnout = (float*)alloc(BN * Dc * 4);
    float* biasbuf = (float*)alloc((size_t)Bc * Hc * Nc * Nc * 4);
    float* ffbuf   = (float*)alloc(BN * FFc * 4);
    float* subjh   = (float*)alloc(BN * Dc * 4);
    float* objhb   = (float*)alloc(BN * Dc * 4);
    // bf16 fragment buffers (sizes in bytes = K*N16*2)
    const size_t fDD  = (size_t)Dc * Dc / 2;          // dwords per 256x256
    const size_t fDF  = (size_t)Dc * FFc / 2;         // dwords per 256x1024 / 1024x256
    unsigned int* qf   = (unsigned int*)alloc(Lc * fDD * 4);
    unsigned int* kf   = (unsigned int*)alloc(Lc * fDD * 4);
    unsigned int* vf   = (unsigned int*)alloc(Lc * fDD * 4);
    unsigned int* of   = (unsigned int*)alloc(Lc * fDD * 4);
    unsigned int* ff1f = (unsigned int*)alloc(Lc * fDF * 4);
    unsigned int* ff2f = (unsigned int*)alloc(Lc * fDF * 4);
    unsigned int* ohf  = (unsigned int*)alloc((size_t)Dc * 160 / 2 * 4);
    unsigned int* sjf  = (unsigned int*)alloc(fDD * 4);
    unsigned int* opf  = (unsigned int*)alloc(fDD * 4);
    unsigned int* ehf  = (unsigned int*)alloc(fDF * 4);
    unsigned int* rhf  = (unsigned int*)alloc(fDF * 4);

    // ---- weight prep ----
    for (int l = 0; l < Lc; ++l) {
        prep_frag_kernel<<<128, 256, 0, stream>>>(q_w + (size_t)l * Dc * Dc,  qf + l * fDD, Dc, Dc, 16);
        prep_frag_kernel<<<128, 256, 0, stream>>>(k_w + (size_t)l * Dc * Dc,  kf + l * fDD, Dc, Dc, 16);
        prep_frag_kernel<<<128, 256, 0, stream>>>(v_w + (size_t)l * Dc * Dc,  vf + l * fDD, Dc, Dc, 16);
        prep_frag_kernel<<<128, 256, 0, stream>>>(o_w + (size_t)l * Dc * Dc,  of + l * fDD, Dc, Dc, 16);
        prep_frag_kernel<<<512, 256, 0, stream>>>(ff_w1 + (size_t)l * Dc * FFc, ff1f + l * fDF, Dc, FFc, 64);
        prep_frag_kernel<<<512, 256, 0, stream>>>(ff_w2 + (size_t)l * FFc * Dc, ff2f + l * fDF, FFc, Dc, 16);
    }
    prep_frag_kernel<<<80,  256, 0, stream>>>(obj_head_w, ohf, Dc, NOBJ, 10);
    prep_frag_kernel<<<128, 256, 0, stream>>>(subj_w, sjf, Dc, Dc, 16);
    prep_frag_kernel<<<128, 256, 0, stream>>>(objp_w, opf, Dc, Dc, 16);
    prep_frag_kernel<<<512, 256, 0, stream>>>(eh_w1, ehf, 4 * Dc, Dc, 16);
    prep_frag_kernel<<<512, 256, 0, stream>>>(rh_w1, rhf, 4 * Dc, Dc, 16);

    // ---- embedding + time MLP ----
    time_mlp_kernel<<<Bc, 256, 0, stream>>>(t_in, time_w1, time_b1, time_w2, time_b2, temb);
    embed_kernel<<<(int)(BN * Dc / 256), 256, 0, stream>>>(obj_t, obj_emb, temb, node_mask, hbuf);

    // ---- transformer layers ----
    const dim3 gemmBlk(128);
    for (int l = 0; l < Lc; ++l) {
        ln_kernel<<<(int)BN, 256, 0, stream>>>(hbuf, xbuf, ln1_g + l * Dc, ln1_b + l * Dc);
        gemm_bf16_kernel<<<dim3(64, 4), gemmBlk, 0, stream>>>(xbuf, qf + l * fDD, q_b + l * Dc, qbuf, nullptr, Dc, 8, Dc, 16, 0);
        gemm_bf16_kernel<<<dim3(64, 4), gemmBlk, 0, stream>>>(xbuf, kf + l * fDD, k_b + l * Dc, kbuf, nullptr, Dc, 8, Dc, 16, 0);
        gemm_bf16_kernel<<<dim3(64, 4), gemmBlk, 0, stream>>>(xbuf, vf + l * fDD, v_b + l * Dc, vbuf, nullptr, Dc, 8, Dc, 16, 0);
        bias_kernel<<<(int)BN, 128, 0, stream>>>(rel_t, rel_emb, eb_w + l * Dc * Hc, eb_b + l * Hc, biasbuf);
        attn_kernel<<<Bc * Hc * Nc, 128, 0, stream>>>(qbuf, kbuf, vbuf, biasbuf, node_mask, edge_mask, attnout);
        gemm_bf16_kernel<<<dim3(64, 4), gemmBlk, 0, stream>>>(attnout, of + l * fDD, o_b + l * Dc, hbuf, hbuf, Dc, 8, Dc, 16, 0);
        ln_kernel<<<(int)BN, 256, 0, stream>>>(hbuf, xbuf, ln2_g + l * Dc, ln2_b + l * Dc);
        gemm_bf16_kernel<<<dim3(64, 16), gemmBlk, 0, stream>>>(xbuf, ff1f + l * fDF, ff_b1 + l * FFc, ffbuf, nullptr, Dc, 8, FFc, 64, 1);
        gemm_bf16_kernel<<<dim3(64, 4), gemmBlk, 0, stream>>>(ffbuf, ff2f + l * fDF, ff_b2 + l * Dc, hbuf, hbuf, FFc, 10, Dc, 16, 0);
        maskmul_kernel<<<(int)(BN * Dc / 256), 256, 0, stream>>>(hbuf, node_mask);
    }

    // ---- heads ----
    float* out_obj  = (float*)d_out;                       // [B,N,150]
    float* out_edge = out_obj + (size_t)Bc * Nc * NOBJ;    // [B,N,N]
    float* out_rel  = out_edge + (size_t)Bc * Nc * Nc;     // [B,N,N,50]

    gemm_bf16_kernel<<<dim3(64, 3), gemmBlk, 0, stream>>>(hbuf, ohf, obj_head_b, out_obj, nullptr, Dc, 8, NOBJ, 10, 0);
    gemm_bf16_kernel<<<dim3(64, 4), gemmBlk, 0, stream>>>(hbuf, sjf, subj_b, subjh, nullptr, Dc, 8, Dc, 16, 0);
    gemm_bf16_kernel<<<dim3(64, 4), gemmBlk, 0, stream>>>(hbuf, opf, objp_b, objhb, nullptr, Dc, 8, Dc, 16, 0);

    pair_head_kernel<<<Bc * Nc * (Nc / 16), 32, 0, stream>>>(
        subjh, objhb, ehf, rhf, eh_b1, rh_b1, eh_w2, eh_b2, rh_w2, rh_b2,
        out_edge, out_rel);
}
